// GNNEncoder_32770600468961
// MI455X (gfx1250) — matrix operand order repaired and checked by reading, not compiled, not run on hardware
//
#include <hip/hip_runtime.h>

// Problem constants (match reference)
#define BB   32
#define NN   2048
#define HH   64
#define KK   16
#define OUTD 128

typedef __bf16 v16bf __attribute__((ext_vector_type(16)));
typedef float  v8f   __attribute__((ext_vector_type(8)));

struct U8 { uint4 a, c; };   // 32 bytes == one v16bf fragment

__device__ __forceinline__ unsigned short f2bfu(float f) {
    unsigned int u = __builtin_bit_cast(unsigned int, f);
    unsigned int r = u + 0x7FFFu + ((u >> 16) & 1u);   // round-to-nearest-even
    return (unsigned short)(r >> 16);
}
__device__ __forceinline__ float bfu2f(unsigned short s) {
    unsigned int u = ((unsigned int)s) << 16;
    return __builtin_bit_cast(float, u);
}
// packed bf16x2 subtract (a - b) on raw dwords
__device__ __forceinline__ unsigned int bfsub2(unsigned int a, unsigned int b) {
    float alo = bfu2f((unsigned short)a), ahi = bfu2f((unsigned short)(a >> 16));
    float blo = bfu2f((unsigned short)b), bhi = bfu2f((unsigned short)(b >> 16));
    return (unsigned int)f2bfu(alo - blo) | ((unsigned int)f2bfu(ahi - bhi) << 16);
}
__device__ __forceinline__ v16bf ldfrag_g(const unsigned short* p) {
    U8 u;
    u.a = ((const uint4*)p)[0];
    u.c = ((const uint4*)p)[1];
    return __builtin_bit_cast(v16bf, u);
}
__device__ __forceinline__ v16bf ldfrag2_lds(const unsigned short* p0,
                                             const unsigned short* p1) {
    U8 u;
    u.a = *(const uint4*)p0;   // elements 0..7  (K run 1)
    u.c = *(const uint4*)p1;   // elements 8..15 (K run 2)
    return __builtin_bit_cast(v16bf, u);
}

// ---------------------------------------------------------------------------
// Convert feature matrix [B,N,D] f32 -> zero-padded bf16 [B,N,64] + sq norms
// ---------------------------------------------------------------------------
__global__ void to_bf16_sq_kernel(const float* __restrict__ in, int D,
                                  unsigned short* __restrict__ xb,
                                  float* __restrict__ sq) {
    int node = blockIdx.x * blockDim.x + threadIdx.x;
    if (node >= BB * NN) return;
    const float* p = in + (size_t)node * D;
    unsigned short* o = xb + ((size_t)node << 6);
    float s = 0.f;
    for (int i = 0; i < HH; ++i) {
        float v = (i < D) ? p[i] : 0.f;
        s += v * v;
        o[i] = f2bfu(v);
    }
    sq[node] = s;
}

// ---------------------------------------------------------------------------
// Weight prep: emit weights directly in WMMA B-fragment order.
// GEMM1 frag buffer (per layer, 8192 bf16): idx = ((nt*4+kb)*32 + lane)*16 + i
//   value = w1_padded[kb*32 + (lane>=16)*16 + i][nt*16 + (lane&15)]
//   (layer0 pad: rows {0..2} <- w1_0 rows 0..2, rows {64..66} <- w1_0 rows 3..5)
// GEMM2 frag buffer (per layer, 4096 bf16): idx = ((nt*2+kb)*32 + lane)*16 + i
// ---------------------------------------------------------------------------
__global__ void prep_weights_kernel(const float* __restrict__ w1_0,
                                    const float* __restrict__ wo_0,
                                    const float* __restrict__ w1_1,
                                    const float* __restrict__ wo_1,
                                    const float* __restrict__ w1_2,
                                    const float* __restrict__ wo_2,
                                    unsigned short* __restrict__ w1f,
                                    unsigned short* __restrict__ wof) {
    int t = blockIdx.x * blockDim.x + threadIdx.x;
    int stride = gridDim.x * blockDim.x;
    for (int t0 = t; t0 < 3 * 8192; t0 += stride) {
        int l    = t0 / 8192;
        int idx  = t0 % 8192;
        int i    = idx & 15;
        int lane = (idx >> 4) & 31;
        int kb   = (idx >> 9) & 3;
        int nt   = (idx >> 11) & 3;
        int r = kb * 32 + ((lane >> 4) << 4) + i;   // 0..127
        int c = nt * 16 + (lane & 15);              // 0..63
        float v = 0.f;
        if (l == 0) {
            if (r < 3)                  v = w1_0[r * 64 + c];
            else if (r >= 64 && r < 67) v = w1_0[(r - 61) * 64 + c];
        } else {
            const float* w = (l == 1) ? w1_1 : w1_2;
            v = w[r * 64 + c];
        }
        w1f[t0] = f2bfu(v);
    }
    for (int t0 = t; t0 < 3 * 4096; t0 += stride) {
        int l    = t0 / 4096;
        int idx  = t0 % 4096;
        int i    = idx & 15;
        int lane = (idx >> 4) & 31;
        int kb   = (idx >> 9) & 1;
        int nt   = (idx >> 10) & 3;
        int r = kb * 32 + ((lane >> 4) << 4) + i;   // 0..63
        int c = nt * 16 + (lane & 15);
        const float* w = (l == 0) ? wo_0 : (l == 1) ? wo_1 : wo_2;
        wof[t0] = f2bfu(w[r * 64 + c]);
    }
}

// ---------------------------------------------------------------------------
// kNN: block = 16 query rows of one batch, 4 waves.
// Gram tiles via v_wmma_f32_16x16x32_bf16 (K=64 -> 2 chained WMMAs),
// d2 = sq_i + sq_j - 2*G, register top-16 per thread, merged at end.
// ---------------------------------------------------------------------------
__global__ __launch_bounds__(128) void knn_kernel(
        const unsigned short* __restrict__ xb, const float* __restrict__ sq,
        int* __restrict__ knn_out) {
    __shared__ unsigned short xr[16][72];     // padded row features (16B-aligned rows)
    __shared__ float sqr[16];
    __shared__ float d2s[16][68];             // padded distance tile (64 cols used)
    __shared__ float topd[128][16];
    __shared__ int   topi[128][16];

    const int tid  = threadIdx.x;
    const int lane = tid & 31;
    const int wave = tid >> 5;
    const int b    = blockIdx.x >> 7;         // 128 row-tiles per batch
    const int row0 = (blockIdx.x & 127) << 4;

    for (int i = tid; i < 16 * 64; i += 128) {
        int r = i >> 6, c = i & 63;
        xr[r][c] = xb[((size_t)(b * NN + row0 + r) << 6) + c];
    }
    if (tid < 16) sqr[tid] = sq[b * NN + row0 + tid];
    __syncthreads();

    // A fragments: row = lane&15; lanes<16 hold K {0..15}, lanes>=16 {16..31}
    const int ar = lane & 15;
    const int k8 = (lane >> 4) << 4;
    v16bf a0 = ldfrag2_lds(&xr[ar][k8],      &xr[ar][8 + k8]);
    v16bf a1 = ldfrag2_lds(&xr[ar][32 + k8], &xr[ar][40 + k8]);

    float dd[16]; int ii[16];
#pragma unroll
    for (int i = 0; i < 16; ++i) { dd[i] = 3.0e38f; ii[i] = 0; }

    const int selRow  = tid >> 3;             // 8 threads per row
    const int selSlot = tid & 7;
    const int colLane = lane & 15;
    const int koff    = (lane >> 4) << 4;     // B frag: lanes>=16 take K+16
    const int rbase   = (lane >> 4) << 3;

    for (int it = 0; it < 32; ++it) {
        const int col = it * 64 + wave * 16 + colLane;
        const unsigned short* cp = xb + ((size_t)(b * NN + col) << 6) + koff;
        v16bf b0 = ldfrag_g(cp);
        v16bf b1 = ldfrag_g(cp + 32);
        if (it < 31)
            __builtin_prefetch(cp + (64 << 6), 0, 1);   // next column tile

        v8f acc = {0.f, 0.f, 0.f, 0.f, 0.f, 0.f, 0.f, 0.f};
        acc = __builtin_amdgcn_wmma_f32_16x16x32_bf16(false, a0, false, b0,
                                                      (short)0, acc, false, false);
        acc = __builtin_amdgcn_wmma_f32_16x16x32_bf16(false, a1, false, b1,
                                                      (short)0, acc, false, false);
        const float sqc = sq[b * NN + col];
#pragma unroll
        for (int v = 0; v < 8; ++v)
            d2s[rbase + v][wave * 16 + colLane] = sqr[rbase + v] + sqc - 2.0f * acc[v];
        __syncthreads();

#pragma unroll
        for (int j = 0; j < 8; ++j) {
            const int c = selSlot * 8 + j;
            float cand = d2s[selRow][c];
            int cidx = it * 64 + c;
            if (cand < dd[15]) {
                dd[15] = cand; ii[15] = cidx;
#pragma unroll
                for (int s = 15; s > 0; --s) {
                    if (dd[s] < dd[s - 1]) {
                        float tf = dd[s]; dd[s] = dd[s - 1]; dd[s - 1] = tf;
                        int   ti = ii[s]; ii[s] = ii[s - 1]; ii[s - 1] = ti;
                    }
                }
            }
        }
        __syncthreads();
    }

#pragma unroll
    for (int i = 0; i < 16; ++i) { topd[tid][i] = dd[i]; topi[tid][i] = ii[i]; }
    __syncthreads();

    if (tid < 16) {
        float fd[16]; int fi[16];
#pragma unroll
        for (int i = 0; i < 16; ++i) { fd[i] = 3.0e38f; fi[i] = 0; }
        for (int t = tid * 8; t < tid * 8 + 8; ++t) {
            for (int i = 0; i < 16; ++i) {
                float cand = topd[t][i]; int cidx = topi[t][i];
                if (cand < fd[15]) {
                    fd[15] = cand; fi[15] = cidx;
#pragma unroll
                    for (int s = 15; s > 0; --s) {
                        if (fd[s] < fd[s - 1]) {
                            float tf = fd[s]; fd[s] = fd[s - 1]; fd[s - 1] = tf;
                            int   ti = fi[s]; fi[s] = fi[s - 1]; fi[s - 1] = ti;
                        }
                    }
                }
            }
        }
        int* op = knn_out + ((size_t)(b * NN + row0 + tid) << 4);
#pragma unroll
        for (int i = 0; i < 16; ++i) op[i] = fi[i];
    }
}

// ---------------------------------------------------------------------------
// Edge MLP: one wave per node. A = edge matrix [16x128] bf16 (K=16 edges ==
// one WMMA row tile). Weights arrive pre-swizzled in B-fragment order, loaded
// as 2x16B per fragment from global (L2-resident). GEMM1 (16 WMMAs) + bias +
// ReLU -> LDS transpose -> GEMM2 (8 WMMAs) + bias -> mean -> h_out[b,n,64].
// ---------------------------------------------------------------------------
__global__ __launch_bounds__(128) void edge_mlp_kernel(
        const unsigned short* __restrict__ xb, const int* __restrict__ knn_in,
        const unsigned short* __restrict__ w1f, const float* __restrict__ b1,
        const unsigned short* __restrict__ wof, const float* __restrict__ bo,
        float* __restrict__ hout) {
    __shared__ unsigned short es[4][16][136];   // edge features per wave
    __shared__ unsigned short ms[4][16][72];    // mid activations (transpose hop)

    const int tid  = threadIdx.x;
    const int lane = tid & 31;
    const int wave = tid >> 5;

    const int g = blockIdx.x * 4 + wave;        // global node
    const int b = g >> 11;                      // / N (N = 2048)
    const int er   = lane >> 1;                 // edge row 0..15 (2 lanes/row)
    const int half = lane & 1;                  // 0: x_i, 1: x_j - x_i
    const int j = knn_in[((size_t)g << 4) + er];
    const unsigned short* xi = xb + ((size_t)g << 6);
    const unsigned short* xj = xb + ((size_t)(b * NN + j) << 6);
    if (half == 0) {
        const uint4* ps = (const uint4*)xi;
        uint4* pd = (uint4*)&es[wave][er][0];
#pragma unroll
        for (int q = 0; q < 8; ++q) pd[q] = ps[q];
    } else {
        const uint4* pj = (const uint4*)xj;
        const uint4* pi = (const uint4*)xi;
        uint4* pd = (uint4*)&es[wave][er][64];
#pragma unroll
        for (int q = 0; q < 8; ++q) {
            uint4 aa = pj[q], bb = pi[q], r;
            r.x = bfsub2(aa.x, bb.x);
            r.y = bfsub2(aa.y, bb.y);
            r.z = bfsub2(aa.z, bb.z);
            r.w = bfsub2(aa.w, bb.w);
            pd[q] = r;
        }
    }
    __syncthreads();

    const int ar = lane & 15;
    const int k8 = (lane >> 4) << 4;
    const int colL  = lane & 15;
    const int rbase = (lane >> 4) << 3;

    v16bf A[4];
#pragma unroll
    for (int kb = 0; kb < 4; ++kb)
        A[kb] = ldfrag2_lds(&es[wave][ar][kb * 32 + k8],
                            &es[wave][ar][kb * 32 + 8 + k8]);

    // GEMM1: [16x128] @ [128x64] + b1, ReLU
#pragma unroll
    for (int nt = 0; nt < 4; ++nt) {
        const int col = nt * 16 + colL;
        float bias = b1[col];
        v8f acc = {bias, bias, bias, bias, bias, bias, bias, bias};
#pragma unroll
        for (int kb = 0; kb < 4; ++kb) {
            v16bf Bf = ldfrag_g(w1f + ((((nt << 2) + kb) << 5) + lane) * 16);
            acc = __builtin_amdgcn_wmma_f32_16x16x32_bf16(false, A[kb], false, Bf,
                                                          (short)0, acc, false, false);
        }
#pragma unroll
        for (int v = 0; v < 8; ++v) {
            float r = acc[v] > 0.f ? acc[v] : 0.f;
            ms[wave][rbase + v][col] = f2bfu(r);
        }
    }
    __syncthreads();

    // GEMM2: [16x64] @ [64x64] + bo, then mean over the 16 edges
    v16bf A2[2];
#pragma unroll
    for (int kb = 0; kb < 2; ++kb)
        A2[kb] = ldfrag2_lds(&ms[wave][ar][kb * 32 + k8],
                             &ms[wave][ar][kb * 32 + 8 + k8]);

    float outv[4];
#pragma unroll
    for (int nt = 0; nt < 4; ++nt) {
        const int col = nt * 16 + colL;
        float bias = bo[col];
        v8f acc = {bias, bias, bias, bias, bias, bias, bias, bias};
#pragma unroll
        for (int kb = 0; kb < 2; ++kb) {
            v16bf Bf = ldfrag_g(wof + ((((nt << 1) + kb) << 5) + lane) * 16);
            acc = __builtin_amdgcn_wmma_f32_16x16x32_bf16(false, A2[kb], false, Bf,
                                                          (short)0, acc, false, false);
        }
        float p = 0.f;
#pragma unroll
        for (int v = 0; v < 8; ++v) p += acc[v];
        p += __shfl_xor(p, 16, 32);             // add rows 8..15 half
        outv[nt] = p * (1.0f / (float)KK);
    }
    if (lane < 16) {
        float* op = hout + ((size_t)g << 6);
#pragma unroll
        for (int nt = 0; nt < 4; ++nt) op[nt * 16 + colL] = outv[nt];
    }
}

// ---------------------------------------------------------------------------
// Global mean pool over N, then tiny 2-layer head (scalar; negligible FLOPs)
// ---------------------------------------------------------------------------
__global__ void pool_kernel(const float* __restrict__ h, float* __restrict__ g) {
    __shared__ float ps[4][64];
    const int b = blockIdx.x;
    const int t = threadIdx.x;                  // 256
    const int c = t & 63;
    const int grp = t >> 6;
    float s = 0.f;
    for (int n = grp; n < NN; n += 4)
        s += h[((size_t)(b * NN + n) << 6) + c];
    ps[grp][c] = s;
    __syncthreads();
    if (t < 64)
        g[b * 64 + t] = (ps[0][t] + ps[1][t] + ps[2][t] + ps[3][t]) * (1.0f / (float)NN);
}

__global__ void head_kernel(const float* __restrict__ g,
                            const float* __restrict__ fw1, const float* __restrict__ fb1,
                            const float* __restrict__ fw2, const float* __restrict__ fb2,
                            float* __restrict__ out) {
    __shared__ float mid[32][64];
    const int t = threadIdx.x;                  // 256
    for (int i = t; i < 32 * 64; i += 256) {
        int r = i >> 6, c = i & 63;
        float s = fb1[c];
        for (int k = 0; k < 64; ++k) s += g[r * 64 + k] * fw1[k * 64 + c];
        mid[r][c] = s > 0.f ? s : 0.f;
    }
    __syncthreads();
    for (int i = t; i < 32 * OUTD; i += 256) {
        int r = i >> 7, o = i & 127;
        float s = fb2[o];
        for (int c = 0; c < 64; ++c) s += mid[r][c] * fw2[c * OUTD + o];
        out[i] = s;
    }
}

// ---------------------------------------------------------------------------
extern "C" void kernel_launch(void* const* d_in, const int* in_sizes, int n_in,
                              void* d_out, int out_size, void* d_ws, size_t ws_size,
                              hipStream_t stream) {
    (void)in_sizes; (void)n_in; (void)out_size; (void)ws_size;
    const float* x   = (const float*)d_in[0];
    const float* w1l[3] = {(const float*)d_in[1], (const float*)d_in[5], (const float*)d_in[9]};
    const float* b1l[3] = {(const float*)d_in[2], (const float*)d_in[6], (const float*)d_in[10]};
    const float* wol[3] = {(const float*)d_in[3], (const float*)d_in[7], (const float*)d_in[11]};
    const float* bol[3] = {(const float*)d_in[4], (const float*)d_in[8], (const float*)d_in[12]};
    const float* fw1 = (const float*)d_in[13];
    const float* fb1 = (const float*)d_in[14];
    const float* fw2 = (const float*)d_in[15];
    const float* fb2 = (const float*)d_in[16];

    // Workspace carve-out (~46 MB total)
    size_t off = 0;
    char* base = (char*)d_ws;
    auto take = [&](size_t bytes) -> void* {
        void* p = base + off;
        off += (bytes + 255) & ~(size_t)255;
        return p;
    };
    unsigned short* xb   = (unsigned short*)take((size_t)BB * NN * HH * 2);
    float*          sq   = (float*)take((size_t)BB * NN * 4);
    int*            knnb = (int*)take((size_t)BB * NN * KK * 4);
    float*          hA   = (float*)take((size_t)BB * NN * HH * 4);
    float*          hB   = (float*)take((size_t)BB * NN * HH * 4);
    unsigned short* w1f  = (unsigned short*)take((size_t)3 * 8192 * 2);
    unsigned short* wof  = (unsigned short*)take((size_t)3 * 4096 * 2);
    float*          gbuf = (float*)take((size_t)BB * HH * 4);

    prep_weights_kernel<<<64, 256, 0, stream>>>(w1l[0], wol[0], w1l[1], wol[1],
                                                w1l[2], wol[2], w1f, wof);

    const float* hin = x;
    int D = 3;
    float* houts[3] = {hA, hB, hA};
    for (int l = 0; l < 3; ++l) {
        to_bf16_sq_kernel<<<(BB * NN + 255) / 256, 256, 0, stream>>>(hin, D, xb, sq);
        knn_kernel<<<BB * NN / 16, 128, 0, stream>>>(xb, sq, knnb);
        edge_mlp_kernel<<<BB * NN / 4, 128, 0, stream>>>(
            xb, knnb, w1f + (size_t)l * 8192, b1l[l],
            wof + (size_t)l * 4096, bol[l], houts[l]);
        hin = houts[l];
        D = HH;
    }
    pool_kernel<<<BB, 256, 0, stream>>>(houts[2], gbuf);
    head_kernel<<<1, 256, 0, stream>>>(gbuf, fw1, fb1, fw2, fb2, (float*)d_out);
}
